// CrossAttention_18210661335346
// MI455X (gfx1250) — matrix operand order repaired and checked by reading, not compiled, not run on hardware
//
#include <hip/hip_runtime.h>

#define BATCH 8
#define CCH   512
#define NSEQ  2048
#define DQK   64

typedef __attribute__((ext_vector_type(16))) _Float16 v16h;
typedef __attribute__((ext_vector_type(8)))  _Float16 v8h;
typedef __attribute__((ext_vector_type(8)))  float    v8f;

// ---- WMMA fragment loaders (16x16x32 f16; contiguous k = h*16+j order, same for A and B) ----

// A matrix 16x32: lane L -> row m=L%16, half h=L/16; element j -> k=h*16+j
__device__ __forceinline__ v16h load_a_frag(const _Float16* __restrict__ base, int ld, int lane) {
  const int m = lane & 15, h = lane >> 4;
  const _Float16* row = base + (size_t)m * ld + (h << 4);
  v16h a;
#pragma unroll
  for (int j = 0; j < 16; ++j) {
    const int k = j;
    a[j] = row[k];
  }
  return a;
}

// B matrix 32x16: lane L -> col n=L%16, half h=L/16; element j -> k=h*16+j.
// Source stored "n-major, k-contiguous": element(k,n) = base[n*ldn + k]
__device__ __forceinline__ v16h load_b_frag(const _Float16* __restrict__ base, int ldn, int lane) {
  const int n = lane & 15, h = lane >> 4;
  const _Float16* col = base + (size_t)n * ldn + (h << 4);
  v16h b;
#pragma unroll
  for (int j = 0; j < 16; ++j) b[j] = col[j];
  return b;
}

// ---- weight fp32 -> f16 ----
__global__ void cvt_f32_to_f16(const float* __restrict__ s, _Float16* __restrict__ d, int n) {
  int i = blockIdx.x * blockDim.x + threadIdx.x;
  if (i < n) d[i] = (_Float16)s[i];
}

// ---- projection GEMM: Out[d,n] = W[d,:] . X[b,:,n] + bias[d] ----
// Each WG: 64 output rows (4 d-tiles) x 128 cols; grid-y = D/64.
// LAYOUT=0: write Out[b][n][D]   (Q and K-transposed)
// LAYOUT=1: write Out[b][d][NSEQ] (V)
template <int D, int LAYOUT>
__global__ __launch_bounds__(256) void proj_wmma(
    const float* __restrict__ X, const _Float16* __restrict__ W,
    const float* __restrict__ bias, _Float16* __restrict__ Out) {
  // transposed stage: [n: 128][c: 32] pitch 40 halves (80B rows, 16B aligned)
  __shared__ _Float16 sX[128][40];
  const int b = blockIdx.z, d0 = blockIdx.y << 6, n0 = blockIdx.x << 7;
  const int tid = threadIdx.x, wave = tid >> 5, lane = tid & 31;
  const int n = lane & 15, h = lane >> 4;
  const float* Xb = X + (size_t)b * CCH * NSEQ;

  v8f acc[4];
#pragma unroll
  for (int dt = 0; dt < 4; ++dt)
#pragma unroll
    for (int r = 0; r < 8; ++r) acc[dt][r] = 0.f;

  const int nn = tid & 127, ccb = (tid >> 7) << 4;  // thread stages 16 contiguous channels

  for (int c0 = 0; c0 < CCH; c0 += 32) {
    __syncthreads();  // protect previous iteration's reads
    float vbuf[16];
#pragma unroll
    for (int i = 0; i < 16; ++i)
      vbuf[i] = Xb[(size_t)(c0 + ccb + i) * NSEQ + n0 + nn];  // coalesced along nn
    v8h p0, p1;
#pragma unroll
    for (int i = 0; i < 8; ++i) { p0[i] = (_Float16)vbuf[i]; p1[i] = (_Float16)vbuf[8 + i]; }
    *(v8h*)&sX[nn][ccb]     = p0;   // two ds_store_b128 per thread
    *(v8h*)&sX[nn][ccb + 8] = p1;
    __syncthreads();

    // B: element(k=c, col n) = sX[wave*16+n][k] -> two contiguous 16B runs per lane
    const _Float16* src = &sX[(wave << 4) + n][h << 4];
    v16h bb;
#pragma unroll
    for (int j = 0; j < 16; ++j) bb[j] = src[j];
#pragma unroll
    for (int dt = 0; dt < 4; ++dt) {
      v16h a = load_a_frag(W + (size_t)(d0 + (dt << 4)) * CCH + c0, CCH, lane);
      acc[dt] = __builtin_amdgcn_wmma_f32_16x16x32_f16(false, a, false, bb, (short)0, acc[dt], false, false);
    }
  }

  const int np = n0 + (wave << 4) + n;
#pragma unroll
  for (int dt = 0; dt < 4; ++dt) {
    const int dbase = d0 + (dt << 4) + (h << 3);
    if (LAYOUT == 0) {
      v8h pk;
#pragma unroll
      for (int r = 0; r < 8; ++r) pk[r] = (_Float16)(acc[dt][r] + bias[dbase + r]);
      *(v8h*)(Out + ((size_t)b * NSEQ + np) * D + dbase) = pk;  // one 16B store
    } else {
#pragma unroll
      for (int r = 0; r < 8; ++r)
        Out[((size_t)b * D + dbase + r) * NSEQ + np] = (_Float16)(acc[dt][r] + bias[dbase + r]);
    }
  }
}

// ---- flash attention: softmax(Q K) V^T, fused with gamma*out + x ----
// grid: (NSEQ/64, BATCH), block 512 = 16 waves.
// wave w: row tile rt=w>>2 (16 query rows), channel slice cs=w&3 (128 channels).
// Tile-wide max (valid shift, wave-uniform) + row sums accumulated by a
// ones-matrix WMMA: denominator tracks the same rescale recurrence as O.
__global__ __launch_bounds__(512) void attn_wmma(
    const _Float16* __restrict__ Q, const _Float16* __restrict__ Kt,
    const _Float16* __restrict__ V, const float* __restrict__ X,
    const float* __restrict__ gamma, float* __restrict__ Out) {
  __shared__ _Float16 sP[16][16 * 40];  // per-wave P-tile transpose scratch
  const int b = blockIdx.y;
  const int tid = threadIdx.x, wave = tid >> 5, lane = tid & 31;
  const int row0 = (blockIdx.x << 6) + ((wave >> 2) << 4);
  const int c0 = (wave & 3) << 7;
  const int n = lane & 15, h = lane >> 4;

  const _Float16* Qb  = Q  + (size_t)b * NSEQ * DQK;
  const _Float16* Ktb = Kt + (size_t)b * NSEQ * DQK;
  const _Float16* Vb  = V  + (size_t)b * CCH * NSEQ;

  // Q A-fragments (d 0..31 and 32..63), resident for the whole kernel
  const v16h aq0 = load_a_frag(Qb + (size_t)row0 * DQK,      DQK, lane);
  const v16h aq1 = load_a_frag(Qb + (size_t)row0 * DQK + 32, DQK, lane);

  v16h bone;  // all-ones B matrix for row-sum accumulation
#pragma unroll
  for (int j = 0; j < 16; ++j) bone[j] = (_Float16)1.0f;

  v8f O[8], Osum;
  float mi = -1e30f;
#pragma unroll
  for (int t = 0; t < 8; ++t)
#pragma unroll
    for (int r = 0; r < 8; ++r) O[t][r] = 0.f;
#pragma unroll
  for (int r = 0; r < 8; ++r) Osum[r] = 0.f;

  _Float16* Pw = sP[wave];

  for (int m0 = 0; m0 < NSEQ; m0 += 32) {
    const _Float16* Kc = Ktb + (size_t)m0 * DQK;
    if (m0 + 32 < NSEQ) {
      __builtin_prefetch(Kc + 32 * DQK, 0, 1);
      __builtin_prefetch(Vb + (size_t)c0 * NSEQ + m0 + 32, 0, 1);
    }

    // energy tile E = Q(16x64) . K(64x32): 2 col sub-tiles x 2 d-halves
    v8f e0 = {0.f,0.f,0.f,0.f,0.f,0.f,0.f,0.f};
    v8f e1 = {0.f,0.f,0.f,0.f,0.f,0.f,0.f,0.f};
    v16h bk;
    bk = load_b_frag(Kc,                 DQK, lane);
    e0 = __builtin_amdgcn_wmma_f32_16x16x32_f16(false, aq0, false, bk, (short)0, e0, false, false);
    bk = load_b_frag(Kc + 32,            DQK, lane);
    e0 = __builtin_amdgcn_wmma_f32_16x16x32_f16(false, aq1, false, bk, (short)0, e0, false, false);
    bk = load_b_frag(Kc + 16 * DQK,      DQK, lane);
    e1 = __builtin_amdgcn_wmma_f32_16x16x32_f16(false, aq0, false, bk, (short)0, e1, false, false);
    bk = load_b_frag(Kc + 16 * DQK + 32, DQK, lane);
    e1 = __builtin_amdgcn_wmma_f32_16x16x32_f16(false, aq1, false, bk, (short)0, e1, false, false);

    // tile max -> single wave-uniform scalar (5 shuffles)
    float cm = fmaxf(e0[0], e1[0]);
#pragma unroll
    for (int r = 1; r < 8; ++r) cm = fmaxf(cm, fmaxf(e0[r], e1[r]));
#pragma unroll
    for (int off = 1; off < 32; off <<= 1) cm = fmaxf(cm, __shfl_xor(cm, off, 32));

    if (cm > mi) {  // uniform branch; rare after the first few chunks
      const float sc = __expf(mi - cm);
      mi = cm;
#pragma unroll
      for (int t = 0; t < 8; ++t)
#pragma unroll
        for (int r = 0; r < 8; ++r) O[t][r] *= sc;
#pragma unroll
      for (int r = 0; r < 8; ++r) Osum[r] *= sc;
    }

#pragma unroll
    for (int r = 0; r < 8; ++r) {
      e0[r] = __expf(e0[r] - mi);
      e1[r] = __expf(e1[r] - mi);
    }

    // P (16x32) -> LDS (D-layout out), reload as A-fragment (in-order LDS per wave)
#pragma unroll
    for (int r = 0; r < 8; ++r) {
      const int m = r + (h << 3);
      Pw[m * 40 + n]      = (_Float16)e0[r];
      Pw[m * 40 + 16 + n] = (_Float16)e1[r];
    }
    const v16h ap = load_a_frag(Pw, 40, lane);

    // denominator: Osum += P . ones  (row sums, same recurrence as O)
    Osum = __builtin_amdgcn_wmma_f32_16x16x32_f16(false, ap, false, bone, (short)0, Osum, false, false);

    // O(16x128 slice) += P(16x32) . V^T(32x128): 8 WMMAs,
    // software-pipelined: B-frag for tile t+1 is in flight while WMMA t issues
    v16h bv = load_b_frag(Vb + (size_t)c0 * NSEQ + m0, NSEQ, lane);
#pragma unroll
    for (int t = 0; t < 8; ++t) {
      v16h bnext;
      if (t < 7) bnext = load_b_frag(Vb + (size_t)(c0 + ((t + 1) << 4)) * NSEQ + m0, NSEQ, lane);
      O[t] = __builtin_amdgcn_wmma_f32_16x16x32_f16(false, ap, false, bv, (short)0, O[t], false, false);
      if (t < 7) bv = bnext;
    }
  }

  // finalize: out[b,c,n] = gamma * (O / l) + x[b,c,n]; vectorized float4 pairs
  const float g = gamma[0];
  float il[8];
#pragma unroll
  for (int r = 0; r < 8; ++r) il[r] = g / Osum[r];
  const float* Xb = X + (size_t)b * CCH * NSEQ;
  float* Ob = Out + (size_t)b * CCH * NSEQ;
#pragma unroll
  for (int t = 0; t < 8; ++t) {
    const int c = c0 + (t << 4) + n;
    const size_t idx0 = (size_t)c * NSEQ + row0 + (h << 3);  // 32B aligned
    const float4 x0 = *(const float4*)(Xb + idx0);
    const float4 x1 = *(const float4*)(Xb + idx0 + 4);
    float4 o0, o1;
    o0.x = O[t][0] * il[0] + x0.x;  o0.y = O[t][1] * il[1] + x0.y;
    o0.z = O[t][2] * il[2] + x0.z;  o0.w = O[t][3] * il[3] + x0.w;
    o1.x = O[t][4] * il[4] + x1.x;  o1.y = O[t][5] * il[5] + x1.y;
    o1.z = O[t][6] * il[6] + x1.z;  o1.w = O[t][7] * il[7] + x1.w;
    *(float4*)(Ob + idx0)     = o0;
    *(float4*)(Ob + idx0 + 4) = o1;
  }
}

extern "C" void kernel_launch(void* const* d_in, const int* in_sizes, int n_in,
                              void* d_out, int out_size, void* d_ws, size_t ws_size,
                              hipStream_t stream) {
  (void)in_sizes; (void)n_in; (void)out_size; (void)ws_size;
  const float* x  = (const float*)d_in[0];
  const float* y  = (const float*)d_in[1];
  const float* Wq = (const float*)d_in[2];
  const float* bq = (const float*)d_in[3];
  const float* Wk = (const float*)d_in[4];
  const float* bk = (const float*)d_in[5];
  const float* Wv = (const float*)d_in[6];
  const float* bv = (const float*)d_in[7];
  const float* gm = (const float*)d_in[8];
  float* out = (float*)d_out;

  // workspace carve (f16): weights 0.6MB, Q 2MB, Kt 2MB, V 16MB  (~20.6MB total)
  char* w = (char*)d_ws;
  size_t off = 0;
  auto carve = [&](size_t elems) {
    _Float16* p = (_Float16*)(w + off);
    off += ((elems * sizeof(_Float16) + 255) / 256) * 256;
    return p;
  };
  _Float16* WqH = carve((size_t)DQK * CCH);
  _Float16* WkH = carve((size_t)DQK * CCH);
  _Float16* WvH = carve((size_t)CCH * CCH);
  _Float16* Qw  = carve((size_t)BATCH * NSEQ * DQK);
  _Float16* Ktw = carve((size_t)BATCH * NSEQ * DQK);
  _Float16* Vw  = carve((size_t)BATCH * CCH * NSEQ);

  cvt_f32_to_f16<<<(DQK * CCH + 255) / 256, 256, 0, stream>>>(Wq, WqH, DQK * CCH);
  cvt_f32_to_f16<<<(DQK * CCH + 255) / 256, 256, 0, stream>>>(Wk, WkH, DQK * CCH);
  cvt_f32_to_f16<<<(CCH * CCH + 255) / 256, 256, 0, stream>>>(Wv, WvH, CCH * CCH);

  proj_wmma<DQK, 0><<<dim3(NSEQ / 128, DQK / 64, BATCH), 256, 0, stream>>>(x, WqH, bq, Qw);
  proj_wmma<DQK, 0><<<dim3(NSEQ / 128, DQK / 64, BATCH), 256, 0, stream>>>(y, WkH, bk, Ktw);
  proj_wmma<CCH, 1><<<dim3(NSEQ / 128, CCH / 64, BATCH), 256, 0, stream>>>(y, WvH, bv, Vw);

  attn_wmma<<<dim3(NSEQ / 64, BATCH), 512, 0, stream>>>(Qw, Ktw, Vw, x, gm, out);
}
